// GCNModel_59785944760955
// MI455X (gfx1250) — matrix operand order repaired and checked by reading, not compiled, not run on hardware
//
#include <hip/hip_runtime.h>

#define N_NODES   50000
#define N_EDGES   800000
#define IN_DIM    128
#define HID       256
#define N_CLASSES 10
#define N_GRAPHS  64
#define BN_EPS    1e-5f

typedef __attribute__((ext_vector_type(16))) __bf16 v16bf;
typedef __attribute__((ext_vector_type(8)))  float  v8f;

__device__ __forceinline__ void atomAddF(float* p, float v) {
    unsafeAtomicAdd(p, v);   // native global_atomic_add_f32 on gfx1250
}

// gfx1250 async DMA: global -> LDS, 16 bytes per lane, tracked by ASYNCcnt
__device__ __forceinline__ void async_copy16_to_lds(unsigned lds_off, const void* gptr) {
    asm volatile("global_load_async_to_lds_b128 %0, %1, off"
                 :: "v"(lds_off), "v"((unsigned long long)(uintptr_t)gptr)
                 : "memory");
}
__device__ __forceinline__ void wait_async0() {
    asm volatile("s_wait_asynccnt 0" ::: "memory");
}

// ---------------------------------------------------------------- utilities
__global__ void zero_f32(float* p, int n) {
    int i = blockIdx.x * blockDim.x + threadIdx.x;
    if (i < n) p[i] = 0.0f;
}

__global__ void deg_count(const int* __restrict__ dst, float* __restrict__ deg) {
    int e = blockIdx.x * blockDim.x + threadIdx.x;
    if (e < N_EDGES) atomAddF(&deg[dst[e]], 1.0f);
}

__global__ void make_dis(float* dis) {     // in-place: deg -> rsqrt(1+deg)
    int i = blockIdx.x * blockDim.x + threadIdx.x;
    if (i < N_NODES) dis[i] = rsqrtf(1.0f + dis[i]);
}

// ------------------------------------------------ pack W[K x 256] (f32) into
// WMMA B-fragment layout: [kc][ntile][lane][16 halfs], k_local = (j&7) + 8*(lane>=16) + 16*(j>=8)
__global__ void pack_w(const float* __restrict__ W, __bf16* __restrict__ Wp, int K) {
    int idx = blockIdx.x * blockDim.x + threadIdx.x;
    if (idx >= K * HID) return;
    int j    =  idx        & 15;
    int lane = (idx >> 4)  & 31;
    int nt   = (idx >> 9)  & 15;
    int kc   =  idx >> 13;
    int k = kc * 32 + (j & 7) + ((lane >> 4) << 3) + ((j >> 3) << 4);
    int n = nt * 16 + (lane & 15);
    Wp[idx] = (__bf16)W[k * HID + n];
}

// ------------------------------------------------ C[N x 256] = A[N x K] @ Wp
// block = 256 threads = 8 waves; block owns 16 rows; wave w owns cols [32w,32w+32)
// A strip staged once per block into LDS via async DMA (removes 8x redundant fetch)
__global__ __launch_bounds__(256) void gemm_bf16(const float* __restrict__ A,
                                                 const __bf16* __restrict__ Wp,
                                                 float* __restrict__ C, int K) {
    __shared__ float Atile[16][260];                    // +4 pad: conflict-free frag reads

    const int tid     = threadIdx.x;
    const int lane    = tid & 31;
    const int wave    = tid >> 5;
    const int rowbase = blockIdx.x * 16;
    const int KC      = K >> 5;
    const int nt0     = wave * 2;

    // ---- stage A[rowbase:rowbase+16, 0:K] into LDS (16B chunks, all lanes active)
    const int chunksPerRow = K >> 2;                    // float4 chunks per row
    for (int c = tid; c < 16 * chunksPerRow; c += 256) {
        int row = c / chunksPerRow;
        int col = (c - row * chunksPerRow) << 2;
        unsigned loff = (unsigned)(uintptr_t)(&Atile[row][col]);
        async_copy16_to_lds(loff, A + (size_t)(rowbase + row) * K + col);
    }
    wait_async0();
    __syncthreads();

    v8f acc0 = {}; v8f acc1 = {};

    const int rl   = lane & 15;                          // A row within strip
    const int koff = (lane >> 4) << 3;                   // +8 K for hi half-wave
    const __bf16* bp0 = Wp + ((nt0 + 0) * 32 + lane) * 16;
    const __bf16* bp1 = Wp + ((nt0 + 1) * 32 + lane) * 16;

    for (int kc = 0; kc < KC; ++kc) {
        // A fragment (ISA 16-bit A 16x32 layout) from LDS, 4x ds_load_b128 + cvt
        const float* ap = &Atile[rl][kc * 32 + koff];
        float4 p0 = *(const float4*)(ap);
        float4 p1 = *(const float4*)(ap + 4);
        float4 p2 = *(const float4*)(ap + 16);
        float4 p3 = *(const float4*)(ap + 20);
        v16bf a;
        a[0]  = (__bf16)p0.x; a[1]  = (__bf16)p0.y; a[2]  = (__bf16)p0.z; a[3]  = (__bf16)p0.w;
        a[4]  = (__bf16)p1.x; a[5]  = (__bf16)p1.y; a[6]  = (__bf16)p1.z; a[7]  = (__bf16)p1.w;
        a[8]  = (__bf16)p2.x; a[9]  = (__bf16)p2.y; a[10] = (__bf16)p2.z; a[11] = (__bf16)p2.w;
        a[12] = (__bf16)p3.x; a[13] = (__bf16)p3.y; a[14] = (__bf16)p3.z; a[15] = (__bf16)p3.w;

        v16bf b0 = *(const v16bf*)(bp0 + (size_t)kc * 8192);
        v16bf b1 = *(const v16bf*)(bp1 + (size_t)kc * 8192);

        acc0 = __builtin_amdgcn_wmma_f32_16x16x32_bf16(false, a, false, b0, (short)0, acc0, false, false);
        acc1 = __builtin_amdgcn_wmma_f32_16x16x32_bf16(false, a, false, b1, (short)0, acc1, false, false);
    }

    // C/D layout: VGPR r -> M=r (lanes 0-15) / M=r+8 (lanes 16-31), N = lane&15
    const int m0   = (lane >> 4) << 3;
    const int ncol = lane & 15;
    float* crow = C + (size_t)rowbase * HID;
    #pragma unroll
    for (int r = 0; r < 8; ++r) {
        crow[(m0 + r) * HID + (nt0 + 0) * 16 + ncol] = acc0[r];
        crow[(m0 + r) * HID + (nt0 + 1) * 16 + ncol] = acc1[r];
    }
}

// ------------------------------------------------ AGG = H * dis^2 + b  (float4 wide)
__global__ void self_init(const float4* __restrict__ H, const float* __restrict__ dis,
                          const float4* __restrict__ b, float4* __restrict__ AGG) {
    int i = blockIdx.x * blockDim.x + threadIdx.x;
    if (i >= N_NODES * HID / 4) return;
    int n  = i >> 6;                     // 64 float4 per node row
    int c4 = i & 63;
    float d  = dis[n];
    float d2 = d * d;
    float4 h = H[i], bb = b[c4], o;
    o.x = h.x * d2 + bb.x; o.y = h.y * d2 + bb.y;
    o.z = h.z * d2 + bb.z; o.w = h.w * d2 + bb.w;
    AGG[i] = o;
}

// ------------------------------------------------ AGG[dst] += H[src] * dis[src]*dis[dst]
// thread -> (edge, 4-feature chunk); a wave covers 128 contiguous floats of one edge
__global__ void edge_agg(const int* __restrict__ src, const int* __restrict__ dst,
                         const float* __restrict__ dis, const float* __restrict__ H,
                         float* __restrict__ AGG) {
    int idx = blockIdx.x * blockDim.x + threadIdx.x;
    int e = idx >> 6;
    if (e >= N_EDGES) return;
    int c = (idx & 63) << 2;
    int s = src[e], d = dst[e];
    float w = dis[s] * dis[d];
    const float4 v = *(const float4*)(H + (size_t)s * HID + c);
    float* o = AGG + (size_t)d * HID + c;
    atomAddF(o + 0, v.x * w);
    atomAddF(o + 1, v.y * w);
    atomAddF(o + 2, v.z * w);
    atomAddF(o + 3, v.w * w);
}

__global__ void relu_ip(float4* H) {
    int i = blockIdx.x * blockDim.x + threadIdx.x;
    if (i >= N_NODES * HID / 4) return;
    float4 h = H[i];
    h.x = fmaxf(h.x, 0.0f); h.y = fmaxf(h.y, 0.0f);
    h.z = fmaxf(h.z, 0.0f); h.w = fmaxf(h.w, 0.0f);
    H[i] = h;
}

// ------------------------------------------------ BN over relu(H): thread = channel
#define BN_ROWS 128
__global__ __launch_bounds__(256) void bn_stats(const float* __restrict__ H,
                                                float* __restrict__ sums,
                                                float* __restrict__ sumsq) {
    int c  = threadIdx.x;
    int r0 = blockIdx.x * BN_ROWS;
    int r1 = min(r0 + BN_ROWS, N_NODES);
    float s = 0.0f, s2 = 0.0f;
    for (int r = r0; r < r1; ++r) {
        float v = fmaxf(H[(size_t)r * HID + c], 0.0f);
        s += v; s2 += v * v;
    }
    atomAddF(&sums[c], s);
    atomAddF(&sumsq[c], s2);
}

__global__ void bn_finalize(const float* sums, const float* sumsq, float* mu, float* rsig) {
    int c = threadIdx.x;
    if (c >= HID) return;
    float m = sums[c] * (1.0f / N_NODES);
    float v = sumsq[c] * (1.0f / N_NODES) - m * m;
    mu[c]   = m;
    rsig[c] = rsqrtf(fmaxf(v, 0.0f) + BN_EPS);
}

__global__ void bn_apply(float4* __restrict__ H, const float4* __restrict__ mu,
                         const float4* __restrict__ rsig, const float4* __restrict__ g,
                         const float4* __restrict__ be) {
    int i = blockIdx.x * blockDim.x + threadIdx.x;
    if (i >= N_NODES * HID / 4) return;
    int c4 = i & 63;
    float4 h = H[i];
    float4 m = mu[c4], r = rsig[c4], gg = g[c4], bb = be[c4];
    h.x = (fmaxf(h.x, 0.0f) - m.x) * r.x * gg.x + bb.x;   // fused ReLU
    h.y = (fmaxf(h.y, 0.0f) - m.y) * r.y * gg.y + bb.y;
    h.z = (fmaxf(h.z, 0.0f) - m.z) * r.z * gg.z + bb.z;
    h.w = (fmaxf(h.w, 0.0f) - m.w) * r.w * gg.w + bb.w;
    H[i] = h;
}

// ------------------------------------------------ pooling (batch is sorted)
#define POOL_ROWS 256
__global__ __launch_bounds__(256) void pool_sum(const float* __restrict__ H,
                                                const int* __restrict__ batch,
                                                float* __restrict__ pool) {
    int c  = threadIdx.x;
    int r0 = blockIdx.x * POOL_ROWS;
    if (r0 >= N_NODES) return;
    int r1 = min(r0 + POOL_ROWS, N_NODES);
    float acc = 0.0f;
    int g = batch[r0];
    for (int r = r0; r < r1; ++r) {
        int gb = batch[r];
        if (gb != g) { atomAddF(&pool[g * HID + c], acc); acc = 0.0f; g = gb; }
        acc += H[(size_t)r * HID + c];
    }
    atomAddF(&pool[g * HID + c], acc);
}

__global__ void count_nodes(const int* __restrict__ batch, float* __restrict__ cnt) {
    int n = blockIdx.x * blockDim.x + threadIdx.x;
    if (n < N_NODES) atomAddF(&cnt[batch[n]], 1.0f);
}

__global__ void final_fc(const float* __restrict__ pool, const float* __restrict__ cnt,
                         const float* __restrict__ Wf, const float* __restrict__ bf,
                         float* __restrict__ out) {
    int idx = blockIdx.x * blockDim.x + threadIdx.x;
    if (idx >= N_GRAPHS * N_CLASSES) return;
    int g = idx / N_CLASSES, c = idx % N_CLASSES;
    float inv = 1.0f / fmaxf(cnt[g], 1.0f);
    float s = bf[c];
    for (int k = 0; k < HID; ++k)
        s += pool[g * HID + k] * inv * Wf[k * N_CLASSES + c];
    out[idx] = s;
}

// ============================================================== host driver
extern "C" void kernel_launch(void* const* d_in, const int* in_sizes, int n_in,
                              void* d_out, int out_size, void* d_ws, size_t ws_size,
                              hipStream_t stream) {
    const float* x     = (const float*)d_in[0];
    const int*   ei    = (const int*)d_in[1];
    const int*   batch = (const int*)d_in[2];
    const float* W1 = (const float*)d_in[3],  *b1 = (const float*)d_in[4];
    const float* W2 = (const float*)d_in[5],  *b2 = (const float*)d_in[6];
    const float* W3 = (const float*)d_in[7],  *b3 = (const float*)d_in[8];
    const float* g1 = (const float*)d_in[9],  *be1 = (const float*)d_in[10];
    const float* g2 = (const float*)d_in[11], *be2 = (const float*)d_in[12];
    const float* Wf = (const float*)d_in[13], *bf = (const float*)d_in[14];
    float* out = (float*)d_out;

    const int* src = ei;
    const int* dst = ei + N_EDGES;

    // ---- workspace layout (256B aligned blocks)
    char* ws = (char*)d_ws;
    size_t off = 0;
    auto alloc = [&](size_t bytes) { void* p = ws + off; off += (bytes + 255) & ~size_t(255); return p; };
    float*  buf0 = (float*)alloc((size_t)N_NODES * HID * 4);
    float*  buf1 = (float*)alloc((size_t)N_NODES * HID * 4);
    __bf16* W1p  = (__bf16*)alloc((size_t)IN_DIM * HID * 2);
    __bf16* W2p  = (__bf16*)alloc((size_t)HID * HID * 2);
    __bf16* W3p  = (__bf16*)alloc((size_t)HID * HID * 2);
    float*  dis  = (float*)alloc(N_NODES * 4);
    float*  bns  = (float*)alloc(HID * 4);
    float*  bnq  = (float*)alloc(HID * 4);
    float*  bmu  = (float*)alloc(HID * 4);
    float*  brs  = (float*)alloc(HID * 4);
    float*  pool = (float*)alloc(N_GRAPHS * HID * 4);
    float*  cnt  = (float*)alloc(N_GRAPHS * 4);

    const int TB = 256;
    const int gElem4 = (N_NODES * HID / 4 + TB - 1) / TB; // float4 elementwise
    const int gEdgeA = (N_EDGES * 64) / TB;               // edge x 64 chunks
    const int gGemm  = N_NODES / 16;                      // 3125 row strips
    const int gBN    = (N_NODES + BN_ROWS - 1) / BN_ROWS;
    const int gPool  = (N_NODES + POOL_ROWS - 1) / POOL_ROWS;

    // degrees -> dis = rsqrt(1+deg)
    zero_f32<<<(N_NODES + TB - 1) / TB, TB, 0, stream>>>(dis, N_NODES);
    deg_count<<<(N_EDGES + TB - 1) / TB, TB, 0, stream>>>(dst, dis);
    make_dis<<<(N_NODES + TB - 1) / TB, TB, 0, stream>>>(dis);

    // pack weights to WMMA fragment layout
    pack_w<<<(IN_DIM * HID + TB - 1) / TB, TB, 0, stream>>>(W1, W1p, IN_DIM);
    pack_w<<<(HID * HID + TB - 1) / TB, TB, 0, stream>>>(W2, W2p, HID);
    pack_w<<<(HID * HID + TB - 1) / TB, TB, 0, stream>>>(W3, W3p, HID);

    // ---- layer 1: relu(gcn(x, W1, b1))
    gemm_bf16<<<gGemm, TB, 0, stream>>>(x, W1p, buf0, IN_DIM);
    self_init<<<gElem4, TB, 0, stream>>>((const float4*)buf0, dis, (const float4*)b1, (float4*)buf1);
    edge_agg<<<gEdgeA, TB, 0, stream>>>(src, dst, dis, buf0, buf1);
    relu_ip<<<gElem4, TB, 0, stream>>>((float4*)buf1);

    // ---- layer 2: bn1(relu(gcn(h, W2, b2)))
    gemm_bf16<<<gGemm, TB, 0, stream>>>(buf1, W2p, buf0, HID);
    self_init<<<gElem4, TB, 0, stream>>>((const float4*)buf0, dis, (const float4*)b2, (float4*)buf1);
    edge_agg<<<gEdgeA, TB, 0, stream>>>(src, dst, dis, buf0, buf1);
    zero_f32<<<1, HID, 0, stream>>>(bns, HID);
    zero_f32<<<1, HID, 0, stream>>>(bnq, HID);
    bn_stats<<<gBN, TB, 0, stream>>>(buf1, bns, bnq);
    bn_finalize<<<1, HID, 0, stream>>>(bns, bnq, bmu, brs);
    bn_apply<<<gElem4, TB, 0, stream>>>((float4*)buf1, (const float4*)bmu, (const float4*)brs,
                                        (const float4*)g1, (const float4*)be1);

    // ---- layer 3: bn2(relu(gcn(h, W3, b3)))
    gemm_bf16<<<gGemm, TB, 0, stream>>>(buf1, W3p, buf0, HID);
    self_init<<<gElem4, TB, 0, stream>>>((const float4*)buf0, dis, (const float4*)b3, (float4*)buf1);
    edge_agg<<<gEdgeA, TB, 0, stream>>>(src, dst, dis, buf0, buf1);
    zero_f32<<<1, HID, 0, stream>>>(bns, HID);
    zero_f32<<<1, HID, 0, stream>>>(bnq, HID);
    bn_stats<<<gBN, TB, 0, stream>>>(buf1, bns, bnq);
    bn_finalize<<<1, HID, 0, stream>>>(bns, bnq, bmu, brs);
    bn_apply<<<gElem4, TB, 0, stream>>>((float4*)buf1, (const float4*)bmu, (const float4*)brs,
                                        (const float4*)g2, (const float4*)be2);

    // ---- global mean pool + classifier
    zero_f32<<<(N_GRAPHS * HID + TB - 1) / TB, TB, 0, stream>>>(pool, N_GRAPHS * HID);
    zero_f32<<<1, N_GRAPHS, 0, stream>>>(cnt, N_GRAPHS);
    pool_sum<<<gPool, TB, 0, stream>>>(buf1, batch, pool);
    count_nodes<<<(N_NODES + TB - 1) / TB, TB, 0, stream>>>(batch, cnt);
    final_fc<<<(N_GRAPHS * N_CLASSES + TB - 1) / TB, TB, 0, stream>>>(pool, cnt, Wf, bf, out);
}
